// MambaDecoderLayer_3504693314041
// MI455X (gfx1250) — compile-verified, matrix operand order checked by reading, not run on hardware
//
#include <hip/hip_runtime.h>
#include <hip/hip_bf16.h>
#include <math.h>

// ---------------- problem constants ----------------
#define B_SZ     2
#define L_SEQ    2048
#define D_MODEL  2048
#define D_INNER  4096
#define D_STATE  128
#define HEADDIM  64
#define NHEADS   64
#define D_CONV   4
#define INTER    8192
#define CONV_DIM (D_INNER + 2 * D_STATE)                 // 4352
#define D_IN_PROJ (2 * D_INNER + 2 * D_STATE + NHEADS)   // 8512
#define ROWS     (B_SZ * L_SEQ)                           // 4096
#define EPSF     1e-5f

typedef __attribute__((ext_vector_type(16))) __bf16 v16bf;
typedef __attribute__((ext_vector_type(8)))  float  v8f;

// ---------------- CDNA5 async global->LDS helpers ----------------
// LDS byte offset of a generic pointer that points into LDS: explicit
// generic -> AS(3) downcast, then ptrtoint of the 32-bit LDS pointer.
__device__ __forceinline__ unsigned lds_byte_off(const void* p) {
    return (unsigned)(uintptr_t)(__attribute__((address_space(3))) const void*)p;
}

// GLOBAL_LOAD_ASYNC_TO_LDS_B128: copies 16B/lane memory->LDS, tracked by
// ASYNCcnt, no VGPR landing zone (ISA 10 / 15.18.3).
__device__ __forceinline__ void async_cp16(unsigned ldsoff, const void* g) {
    unsigned long long ga = (unsigned long long)g;
    asm volatile("global_load_async_to_lds_b128 %0, %1, off"
                 :: "v"(ldsoff), "v"(ga) : "memory");
}

// ASYNCcnt completes in order -> waiting <= N keeps the newest N copies in
// flight while guaranteeing everything older has landed in LDS.
__device__ __forceinline__ void wait_asynccnt0() {
    asm volatile("s_wait_asynccnt 0x0" ::: "memory");
}
__device__ __forceinline__ void wait_asynccnt2() {
    asm volatile("s_wait_asynccnt 0x2" ::: "memory");
}
__device__ __forceinline__ void wait_asynccnt4() {
    asm volatile("s_wait_asynccnt 0x4" ::: "memory");
}

// ---------------- small elementwise helpers ----------------
__device__ __forceinline__ float silu_f(float x) { return x / (1.0f + __expf(-x)); }

__global__ __launch_bounds__(256) void k_f32_to_bf16(const float* __restrict__ in,
                                                     __bf16* __restrict__ out, size_t n) {
    size_t i = (size_t)blockIdx.x * 256 + threadIdx.x;
    if (i < n) out[i] = (__bf16)in[i];
}

// out[row] = rmsnorm(x[row]) * w   (f32 -> bf16)
__global__ __launch_bounds__(256) void k_rmsnorm_bf16(const float* __restrict__ x,
                                                      const float* __restrict__ w,
                                                      __bf16* __restrict__ out, int D) {
    __shared__ float red[8];
    const int row = blockIdx.x;
    const float* xr = x + (size_t)row * D;
    float ss = 0.0f;
    for (int i = threadIdx.x; i < D; i += 256) { float v = xr[i]; ss += v * v; }
    for (int o = 16; o > 0; o >>= 1) ss += __shfl_xor(ss, o, 32);
    if ((threadIdx.x & 31) == 0) red[threadIdx.x >> 5] = ss;
    __syncthreads();
    float tot = 0.0f;
    #pragma unroll
    for (int i = 0; i < 8; ++i) tot += red[i];
    const float sc = rsqrtf(tot / (float)D + EPSF);
    for (int i = threadIdx.x; i < D; i += 256)
        out[(size_t)row * D + i] = (__bf16)(xr[i] * sc * w[i]);
}

// ---------------- WMMA GEMM: C[M,N] = A[M,K] * W[N,K]^T  (bf16 in, f32 out) ----------
// Double-buffered LDS; global->LDS staging via global_load_async_to_lds_b128
// pipelined one K-tile ahead (wait_asynccnt keeps the new tile in flight).
#define BM 128
#define BN 128
#define BK 32
#define LDT 40  // padded LDS row stride (elements), keeps 16B alignment

__global__ __launch_bounds__(256) void k_gemm_bf16nt(const __bf16* __restrict__ A,
                                                     const __bf16* __restrict__ W,
                                                     float* __restrict__ C,
                                                     int M, int N, int K) {
    __shared__ __align__(16) __bf16 sA[2][BM * LDT];
    __shared__ __align__(16) __bf16 sB[2][BN * LDT];

    const int tid   = threadIdx.x;
    const int lane  = tid & 31;
    const int wave  = tid >> 5;      // 0..7 (wave32)
    const int waveM = wave >> 2;     // 0..1 -> 64 rows each
    const int waveN = wave & 3;      // 0..3 -> 32 cols each
    const int blockM = blockIdx.y * BM;
    const int blockN = blockIdx.x * BN;
    const bool fullN = (blockN + BN) <= N;  // uniform per block: deterministic
                                            // per-wave async issue count

    const int r16 = lane & 15;       // row / col within a 16x16 tile
    const int kh  = lane >> 4;       // K-half selector per ISA 7.12.2 layout

    v8f acc[4][2] = {};
    union Frag { v16bf v; uint4 q[2]; };

    // stage one 128x32 A-tile + 128x32 B-tile into LDS buffer `buf`
    auto stage = [&](int buf, int k0) {
        #pragma unroll
        for (int it = 0; it < 2; ++it) {
            int chunk = tid + it * 256;          // 0..511
            int row = chunk >> 2;                // 0..127
            int col = (chunk & 3) * 8;           // 0,8,16,24
            async_cp16(lds_byte_off(&sA[buf][row * LDT + col]),
                       A + (size_t)(blockM + row) * K + k0 + col);
            if (fullN) {
                async_cp16(lds_byte_off(&sB[buf][row * LDT + col]),
                           W + (size_t)(blockN + row) * K + k0 + col);
            } else {
                // ragged last block column: sync staging with zero-fill so the
                // per-wave ASYNCcnt stays exactly 2 per stage
                int n = blockN + row;
                uint4 bv = {0u, 0u, 0u, 0u};
                if (n < N) bv = *(const uint4*)(W + (size_t)n * K + k0 + col);
                *(uint4*)(&sB[buf][row * LDT + col]) = bv;
            }
        }
    };

    stage(0, 0);
    int buf = 0;
    for (int k0 = 0; k0 < K; k0 += BK) {
        const bool more = (k0 + BK) < K;
        if (more) stage(buf ^ 1, k0 + BK);           // issue next tile (in flight)
        if (k0 + 2 * BK < K) {                       // cache prefetch 2 tiles out
            int pr = tid & 127;
            __builtin_prefetch(A + (size_t)(blockM + pr) * K + k0 + 2 * BK, 0, 1);
        }
        // drain only the previous tile's copies; the just-issued ones keep flying
        if (more) { if (fullN) wait_asynccnt4(); else wait_asynccnt2(); }
        else      { wait_asynccnt0(); }
        __syncthreads();

        // fragment gathers per ISA 16-bit 16x32 layout:
        // lanes 0-15: K=[0..7],[16..23]; lanes 16-31: K=[8..15],[24..31]
        const __bf16* pA = &sA[buf][0];
        const __bf16* pB = &sB[buf][0];
        Frag fa[4], fb[2];
        #pragma unroll
        for (int mt = 0; mt < 4; ++mt) {
            int r = waveM * 64 + mt * 16 + r16;
            fa[mt].q[0] = *(const uint4*)(pA + r * LDT + kh * 8);
            fa[mt].q[1] = *(const uint4*)(pA + r * LDT + 16 + kh * 8);
        }
        #pragma unroll
        for (int nt = 0; nt < 2; ++nt) {
            int r = waveN * 32 + nt * 16 + r16;
            fb[nt].q[0] = *(const uint4*)(pB + r * LDT + kh * 8);
            fb[nt].q[1] = *(const uint4*)(pB + r * LDT + 16 + kh * 8);
        }
        #pragma unroll
        for (int mt = 0; mt < 4; ++mt)
            #pragma unroll
            for (int nt = 0; nt < 2; ++nt)
                acc[mt][nt] = __builtin_amdgcn_wmma_f32_16x16x32_bf16(
                    false, fa[mt].v, false, fb[nt].v, (short)0, acc[mt][nt],
                    false, false);
        __syncthreads();   // all waves done reading `buf` before it is restaged
        buf ^= 1;
    }

    // store: C/D layout -> VGPR i: lanes 0-15 M=i, lanes 16-31 M=i+8; col = lane&15
    #pragma unroll
    for (int mt = 0; mt < 4; ++mt) {
        #pragma unroll
        for (int nt = 0; nt < 2; ++nt) {
            int rowb = blockM + waveM * 64 + mt * 16 + 8 * kh;
            int col  = blockN + waveN * 32 + nt * 16 + r16;
            if (col < N) {
                float* cp = C + (size_t)rowb * N + col;
                #pragma unroll
                for (int i = 0; i < 8; ++i) cp[(size_t)i * N] = acc[mt][nt][i];
            }
        }
    }
}

// ---------------- causal conv1d (4-tap, depthwise) + SiLU ----------------
__global__ __launch_bounds__(256) void k_conv_silu(const float* __restrict__ zx,
                                                   const float* __restrict__ cw,
                                                   const float* __restrict__ cb,
                                                   float* __restrict__ xBC) {
    size_t idx = (size_t)blockIdx.x * 256 + threadIdx.x;
    const size_t total = (size_t)ROWS * CONV_DIM;
    if (idx >= total) return;
    int c = (int)(idx % CONV_DIM);
    int r = (int)(idx / CONV_DIM);       // b*L + l
    int l = r % L_SEQ;
    float acc = cb[c];
    #pragma unroll
    for (int t = 0; t < D_CONV; ++t) {
        int lt = l - (D_CONV - 1) + t;
        if (lt >= 0)
            acc += cw[c * D_CONV + t] *
                   zx[(size_t)(r - l + lt) * D_IN_PROJ + D_INNER + c];
    }
    xBC[idx] = silu_f(acc);
}

__global__ __launch_bounds__(256) void k_dt_softplus(const float* __restrict__ zx,
                                                     const float* __restrict__ dt_bias,
                                                     float* __restrict__ dtb) {
    size_t idx = (size_t)blockIdx.x * 256 + threadIdx.x;
    if (idx >= (size_t)ROWS * NHEADS) return;
    int h = (int)(idx % NHEADS);
    size_t r = idx / NHEADS;
    float v = zx[r * D_IN_PROJ + D_INNER + CONV_DIM + h] + dt_bias[h];
    dtb[idx] = (v > 20.0f) ? v : log1pf(__expf(v));
}

// ---------------- sequential SSM scan: one block per (batch, head) --------------
__global__ __launch_bounds__(256) void k_scan(const float* __restrict__ xBC,
                                              const float* __restrict__ dtb,
                                              const float* __restrict__ A_log,
                                              const float* __restrict__ D_param,
                                              float* __restrict__ y) {
    __shared__ float sB[D_STATE];
    __shared__ float sC[D_STATE];
    const int b = blockIdx.x / NHEADS;
    const int h = blockIdx.x % NHEADS;
    const int t = threadIdx.x;
    const int p  = t >> 2;               // 0..63 head dim element
    const int ng = t & 3;                // 4 lanes split D_STATE
    const float Ah = -__expf(A_log[h]);
    const float Dh = D_param[h];

    float state[32];
    #pragma unroll
    for (int j = 0; j < 32; ++j) state[j] = 0.0f;

    for (int l = 0; l < L_SEQ; ++l) {
        const size_t r = (size_t)b * L_SEQ + l;
        if (t < D_STATE)            sB[t] = xBC[r * CONV_DIM + D_INNER + t];
        else if (t < 2 * D_STATE)   sC[t - D_STATE] =
                 xBC[r * CONV_DIM + D_INNER + D_STATE + (t - D_STATE)];
        __syncthreads();

        const float dt = dtb[r * NHEADS + h];
        const float dA = __expf(dt * Ah);
        const float x  = xBC[r * CONV_DIM + h * HEADDIM + p];
        const float coef = dt * x;

        float partial = 0.0f;
        #pragma unroll
        for (int j = 0; j < 32; ++j) {
            int n = ng * 32 + j;
            float s = state[j] * dA + coef * sB[n];
            state[j] = s;
            partial += s * sC[n];
        }
        partial += __shfl_xor(partial, 1, 32);
        partial += __shfl_xor(partial, 2, 32);
        if (ng == 0)
            y[r * D_INNER + h * HEADDIM + p] = partial + Dh * x;
        __syncthreads();
    }
}

// ---------------- gated RMSNorm: out = rmsnorm(y * silu(z)) * w  (bf16) ----------
__global__ __launch_bounds__(256) void k_gated_norm(const float* __restrict__ y,
                                                    const float* __restrict__ zx,
                                                    const float* __restrict__ w,
                                                    __bf16* __restrict__ out) {
    __shared__ float red[8];
    const int row = blockIdx.x;
    const float* yr = y + (size_t)row * D_INNER;
    const float* zr = zx + (size_t)row * D_IN_PROJ;   // z = first D_INNER cols
    float ss = 0.0f;
    for (int i = threadIdx.x; i < D_INNER; i += 256) {
        float v = yr[i] * silu_f(zr[i]);
        ss += v * v;
    }
    for (int o = 16; o > 0; o >>= 1) ss += __shfl_xor(ss, o, 32);
    if ((threadIdx.x & 31) == 0) red[threadIdx.x >> 5] = ss;
    __syncthreads();
    float tot = 0.0f;
    #pragma unroll
    for (int i = 0; i < 8; ++i) tot += red[i];
    const float sc = rsqrtf(tot / (float)D_INNER + EPSF);
    for (int i = threadIdx.x; i < D_INNER; i += 256) {
        float v = yr[i] * silu_f(zr[i]);
        out[(size_t)row * D_INNER + i] = (__bf16)(v * sc * w[i]);
    }
}

// hs = resid + mamba (f32) ; h2 = rmsnorm(hs)*ln2 (bf16)
__global__ __launch_bounds__(256) void k_resid_norm(const float* __restrict__ resid,
                                                    const float* __restrict__ mamba,
                                                    const float* __restrict__ w,
                                                    float* __restrict__ hs,
                                                    __bf16* __restrict__ h2) {
    __shared__ float red[8];
    const int row = blockIdx.x;
    float ss = 0.0f;
    for (int i = threadIdx.x; i < D_MODEL; i += 256) {
        size_t o = (size_t)row * D_MODEL + i;
        float v = resid[o] + mamba[o];
        hs[o] = v;
        ss += v * v;
    }
    for (int o = 16; o > 0; o >>= 1) ss += __shfl_xor(ss, o, 32);
    if ((threadIdx.x & 31) == 0) red[threadIdx.x >> 5] = ss;
    __syncthreads();
    float tot = 0.0f;
    #pragma unroll
    for (int i = 0; i < 8; ++i) tot += red[i];
    const float sc = rsqrtf(tot / (float)D_MODEL + EPSF);
    for (int i = threadIdx.x; i < D_MODEL; i += 256) {
        size_t o = (size_t)row * D_MODEL + i;
        h2[o] = (__bf16)(hs[o] * sc * w[i]);
    }
}

__global__ __launch_bounds__(256) void k_silu_mul(const float* __restrict__ g,
                                                  const float* __restrict__ u,
                                                  __bf16* __restrict__ act, size_t n) {
    size_t i = (size_t)blockIdx.x * 256 + threadIdx.x;
    if (i < n) act[i] = (__bf16)(silu_f(g[i]) * u[i]);
}

__global__ __launch_bounds__(256) void k_add(const float* __restrict__ a,
                                             const float* __restrict__ b,
                                             float* __restrict__ out, size_t n) {
    size_t i = (size_t)blockIdx.x * 256 + threadIdx.x;
    if (i < n) out[i] = a[i] + b[i];
}

// ---------------- host orchestration ----------------
static inline size_t bump(size_t& off, size_t bytes) {
    size_t o = off;
    off = (off + bytes + 255) & ~(size_t)255;
    return o;
}

extern "C" void kernel_launch(void* const* d_in, const int* in_sizes, int n_in,
                              void* d_out, int out_size, void* d_ws, size_t ws_size,
                              hipStream_t stream) {
    (void)in_sizes; (void)n_in; (void)out_size; (void)ws_size;
    const float* hidden  = (const float*)d_in[0];
    const float* inW     = (const float*)d_in[1];
    const float* convW   = (const float*)d_in[2];
    const float* convB   = (const float*)d_in[3];
    const float* dtBias  = (const float*)d_in[4];
    const float* Alog    = (const float*)d_in[5];
    const float* Dp      = (const float*)d_in[6];
    const float* normW   = (const float*)d_in[7];
    const float* outW    = (const float*)d_in[8];
    const float* ln1     = (const float*)d_in[9];
    const float* ln2     = (const float*)d_in[10];
    const float* gateW   = (const float*)d_in[11];
    const float* upW     = (const float*)d_in[12];
    const float* downW   = (const float*)d_in[13];
    float* outp = (float*)d_out;
    char* ws = (char*)d_ws;

    // persistent region
    size_t off = 0;
    __bf16* wIn   = (__bf16*)(ws + bump(off, (size_t)D_IN_PROJ * D_MODEL * 2));
    __bf16* wOut  = (__bf16*)(ws + bump(off, (size_t)D_MODEL * D_INNER * 2));
    __bf16* wGate = (__bf16*)(ws + bump(off, (size_t)INTER * D_MODEL * 2));
    __bf16* wUp   = (__bf16*)(ws + bump(off, (size_t)INTER * D_MODEL * 2));
    __bf16* wDown = (__bf16*)(ws + bump(off, (size_t)D_MODEL * INTER * 2));
    __bf16* hbf   = (__bf16*)(ws + bump(off, (size_t)ROWS * D_MODEL * 2));
    float*  hsB   = (float*) (ws + bump(off, (size_t)ROWS * D_MODEL * 4));
    __bf16* h2bf  = (__bf16*)(ws + bump(off, (size_t)ROWS * D_MODEL * 2));
    const size_t pool = off;

    // phase 1 scratch (Mamba)
    size_t o1 = pool;
    float*  zx    = (float*) (ws + bump(o1, (size_t)ROWS * D_IN_PROJ * 4));
    float*  xBC   = (float*) (ws + bump(o1, (size_t)ROWS * CONV_DIM * 4));
    float*  dtb   = (float*) (ws + bump(o1, (size_t)ROWS * NHEADS * 4));
    float*  yB    = (float*) (ws + bump(o1, (size_t)ROWS * D_INNER * 4));
    __bf16* ybf   = (__bf16*)(ws + bump(o1, (size_t)ROWS * D_INNER * 2));
    float*  mamba = (float*) (ws + bump(o1, (size_t)ROWS * D_MODEL * 4));
    // phase 2 scratch (MLP) overlays phase 1
    size_t o2 = pool;
    float*  gB    = (float*) (ws + bump(o2, (size_t)ROWS * INTER * 4));
    float*  uB    = (float*) (ws + bump(o2, (size_t)ROWS * INTER * 4));
    __bf16* actB  = (__bf16*)(ws + bump(o2, (size_t)ROWS * INTER * 2));
    float*  mlpB  = (float*) (ws + bump(o2, (size_t)ROWS * D_MODEL * 4));

    auto blk = [](size_t n) { return (unsigned)((n + 255) / 256); };

    // --- weight conversion f32 -> bf16 ---
    size_t nIn = (size_t)D_IN_PROJ * D_MODEL, nOut = (size_t)D_MODEL * D_INNER,
           nMG = (size_t)INTER * D_MODEL;
    k_f32_to_bf16<<<blk(nIn), 256, 0, stream>>>(inW,   wIn,   nIn);
    k_f32_to_bf16<<<blk(nOut), 256, 0, stream>>>(outW, wOut,  nOut);
    k_f32_to_bf16<<<blk(nMG), 256, 0, stream>>>(gateW, wGate, nMG);
    k_f32_to_bf16<<<blk(nMG), 256, 0, stream>>>(upW,   wUp,   nMG);
    k_f32_to_bf16<<<blk(nMG), 256, 0, stream>>>(downW, wDown, nMG);

    // --- ln1 -> bf16 activations ---
    k_rmsnorm_bf16<<<ROWS, 256, 0, stream>>>(hidden, ln1, hbf, D_MODEL);

    // --- in_proj GEMM ---
    {
        dim3 g((D_IN_PROJ + BN - 1) / BN, ROWS / BM);
        k_gemm_bf16nt<<<g, 256, 0, stream>>>(hbf, wIn, zx, ROWS, D_IN_PROJ, D_MODEL);
    }

    // --- conv + silu, dt softplus, scan ---
    k_conv_silu<<<blk((size_t)ROWS * CONV_DIM), 256, 0, stream>>>(zx, convW, convB, xBC);
    k_dt_softplus<<<blk((size_t)ROWS * NHEADS), 256, 0, stream>>>(zx, dtBias, dtb);
    k_scan<<<B_SZ * NHEADS, 256, 0, stream>>>(xBC, dtb, Alog, Dp, yB);

    // --- gated norm + out_proj ---
    k_gated_norm<<<ROWS, 256, 0, stream>>>(yB, zx, normW, ybf);
    {
        dim3 g((D_MODEL + BN - 1) / BN, ROWS / BM);
        k_gemm_bf16nt<<<g, 256, 0, stream>>>(ybf, wOut, mamba, ROWS, D_MODEL, D_INNER);
    }

    // --- residual + ln2 ---
    k_resid_norm<<<ROWS, 256, 0, stream>>>(hidden, mamba, ln2, hsB, h2bf);

    // --- MLP ---
    {
        dim3 g((INTER + BN - 1) / BN, ROWS / BM);
        k_gemm_bf16nt<<<g, 256, 0, stream>>>(h2bf, wGate, gB, ROWS, INTER, D_MODEL);
        k_gemm_bf16nt<<<g, 256, 0, stream>>>(h2bf, wUp,   uB, ROWS, INTER, D_MODEL);
    }
    k_silu_mul<<<blk((size_t)ROWS * INTER), 256, 0, stream>>>(gB, uB, actB,
                                                             (size_t)ROWS * INTER);
    {
        dim3 g((D_MODEL + BN - 1) / BN, ROWS / BM);
        k_gemm_bf16nt<<<g, 256, 0, stream>>>(actB, wDown, mlpB, ROWS, D_MODEL, INTER);
    }
    k_add<<<blk((size_t)ROWS * D_MODEL), 256, 0, stream>>>(hsB, mlpB, outp,
                                                           (size_t)ROWS * D_MODEL);
}
